// DotProductAttention_67482526154827
// MI455X (gfx1250) — matrix operand order repaired and checked by reading, not compiled, not run on hardware
//
#include <hip/hip_runtime.h>
#include <hip/hip_bf16.h>
#include <math.h>

// ---------------------------------------------------------------------------
// DotProductAttention for MI455X (gfx1250, wave32, WMMA bf16 path)
//   out[8192,128] then attention[8192,8192] concatenated in d_out (fp32).
// Two-pass flash scheme: pass 1 row stats only, pass 2 recompute + single
// NT write of the 256 MB attention matrix + P@V WMMA accumulation.
// ---------------------------------------------------------------------------

typedef __attribute__((ext_vector_type(16))) __bf16 v16bf;
typedef __attribute__((ext_vector_type(8)))  __bf16 v8bf;
typedef __attribute__((ext_vector_type(8)))  float  v8f;
typedef __attribute__((ext_vector_type(4)))  int    v4i;

// Pointer-to-vector typedefs with the address space bound to the pointee —
// required for the async-to-LDS builtin (param 1 is v4i addrspace(1)*).
typedef __attribute__((address_space(1))) v4i* as1_v4i_ptr;
typedef __attribute__((address_space(3))) v4i* as3_v4i_ptr;

#define WMMA_BF16(A, B, C) \
  __builtin_amdgcn_wmma_f32_16x16x32_bf16(false, (A), false, (B), (short)0, (C), false, false)

constexpr int N_TOK   = 8192;
constexpr int D_MODEL = 1024;
constexpr int DKV     = 128;

// Native f32 -> bf16 (fptrunc, RNE) — backend emits v_cvt ops.
__device__ __forceinline__ __bf16 f2bf(float f) { return (__bf16)f; }
__device__ __forceinline__ unsigned short f2bf_u(float f) {
  return __builtin_bit_cast(unsigned short, (__bf16)f);
}

// Assemble a v16bf A/B operand from two contiguous 8-element bf16 runs.
__device__ __forceinline__ v16bf ld16(const unsigned short* p0, const unsigned short* p1) {
  v8bf a = *reinterpret_cast<const v8bf*>(p0);
  v8bf b = *reinterpret_cast<const v8bf*>(p1);
  v16bf r;
#pragma unroll
  for (int i = 0; i < 8; ++i) { r[i] = a[i]; r[8 + i] = b[i]; }
  return r;
}

__device__ __forceinline__ v8f v8f_zero() { v8f z = {}; return z; }

// ---------------------------------------------------------------------------
// Async memory -> LDS staging (GLOBAL_LOAD_ASYNC_TO_LDS_B128, ASYNCcnt).
// Guarded so the kernel still compiles on toolchains without the builtin.
// ---------------------------------------------------------------------------
#if defined(__HIP_DEVICE_COMPILE__) && __has_builtin(__builtin_amdgcn_global_load_async_to_lds_b128)
#define HAVE_ASYNC_LDS 1
#else
#define HAVE_ASYNC_LDS 0
#endif

__device__ __forceinline__ void stage16B(const unsigned short* g, unsigned short* l) {
#if HAVE_ASYNC_LDS
  __builtin_amdgcn_global_load_async_to_lds_b128(
      (as1_v4i_ptr)(unsigned short*)g, (as3_v4i_ptr)l, 0, 0);
#else
  *reinterpret_cast<uint4*>(l) = *reinterpret_cast<const uint4*>(g);
#endif
}

__device__ __forceinline__ void async_stage_fence() {
#if HAVE_ASYNC_LDS
#if __has_builtin(__builtin_amdgcn_s_wait_asynccnt)
  __builtin_amdgcn_s_wait_asynccnt(0);
#else
  asm volatile("s_wait_asynccnt 0" ::: "memory");
#endif
#endif
}

// ---------------------------------------------------------------------------
// Kernel 1: projections.  X[8192,1024] @ W[1024,128] -> bf16 out[8192,128].
// blockIdx.y selects Q/K/V.  8 waves/block, each wave owns a 16-row tile.
// ---------------------------------------------------------------------------
__global__ __launch_bounds__(256) void proj_kernel(
    const float* __restrict__ X,
    const float* __restrict__ Wq, const float* __restrict__ Wk,
    const float* __restrict__ Wv,
    unsigned short* __restrict__ Qb, unsigned short* __restrict__ Kb,
    unsigned short* __restrict__ Vb)
{
  const float* W; unsigned short* Out;
  if (blockIdx.y == 0)      { W = Wq; Out = Qb; }
  else if (blockIdx.y == 1) { W = Wk; Out = Kb; }
  else                      { W = Wv; Out = Vb; }

  // W chunk staged TRANSPOSED: Wt[col][k] so B-tile reads are contiguous.
  __shared__ __align__(16) unsigned short Wt[DKV][32];

  const int t    = threadIdx.x;
  const int wave = t >> 5, lane = t & 31;
  const int half = lane >> 4, lc = lane & 15;
  const int xr0  = blockIdx.x * 128 + wave * 16;

  v8f acc[8];
#pragma unroll
  for (int i = 0; i < 8; ++i) acc[i] = v8f_zero();

  for (int kc = 0; kc < D_MODEL / 32; ++kc) {
    const int kb0 = kc * 32;
    __syncthreads();
#pragma unroll
    for (int i = 0; i < 16; ++i) {          // 4096 elems / 256 threads
      int idx = t + 256 * i;
      int r = idx >> 7, c = idx & 127;      // coalesced read of W rows
      Wt[c][r] = f2bf_u(W[(size_t)(kb0 + r) * DKV + c]);
    }
    __syncthreads();

    // A-tile (16x32 bf16): lane<16 holds K=0..7,16..23 ; lane>=16 K=8..15,24..31
    const float* xrow = X + (size_t)(xr0 + lc) * D_MODEL;
    v16bf a;
#pragma unroll
    for (int j = 0; j < 8; ++j) {
      a[j]     = f2bf(xrow[kb0 + half * 8 + j]);
      a[8 + j] = f2bf(xrow[kb0 + 16 + half * 8 + j]);
    }

#pragma unroll
    for (int ct = 0; ct < 8; ++ct) {
      const unsigned short* bp = &Wt[ct * 16 + lc][half * 8];
      acc[ct] = WMMA_BF16(a, ld16(bp, bp + 16), acc[ct]);
    }
  }

  // C layout: VGPR r -> row r (lanes 0-15) / r+8 (lanes 16-31), col = lane%16
#pragma unroll
  for (int ct = 0; ct < 8; ++ct)
#pragma unroll
    for (int r = 0; r < 8; ++r) {
      int row = xr0 + r + 8 * half;
      Out[(size_t)row * DKV + ct * 16 + lc] = f2bf_u(acc[ct][r]);
    }
}

// ---------------------------------------------------------------------------
// Kernel 2: pass 1 — per-row softmax stats (max, sum of exp) over all 8192
// columns via streaming bf16 WMMA QK^T tiles. No N x N traffic.
// ---------------------------------------------------------------------------
__global__ __launch_bounds__(256) void stats_kernel(
    const unsigned short* __restrict__ Qb, const unsigned short* __restrict__ Kb,
    float* __restrict__ mrow, float* __restrict__ drow)
{
  __shared__ __align__(16) unsigned short Kc[32][DKV];

  const int t    = threadIdx.x;
  const int wave = t >> 5, lane = t & 31;
  const int half = lane >> 4, lc = lane & 15;
  const int q0   = blockIdx.x * 128 + wave * 16;
  const float scale = 0.08838834764831845f;   // 1/sqrt(128)

  // Q A-tiles for the whole d_k, kept in registers for all 256 chunks.
  v16bf aq[4];
#pragma unroll
  for (int kk = 0; kk < 4; ++kk) {
    const unsigned short* p = Qb + (size_t)(q0 + lc) * DKV + kk * 32 + half * 8;
    aq[kk] = ld16(p, p + 16);
  }

  float m[8], s[8];
#pragma unroll
  for (int r = 0; r < 8; ++r) { m[r] = -3.0e38f; s[r] = 0.f; }

  for (int cb = 0; cb < N_TOK / 32; ++cb) {
    const int cb0 = cb * 32;
    __syncthreads();
    {
      const unsigned short* src = Kb + (size_t)cb0 * DKV;
      unsigned short* dst = &Kc[0][0];
      stage16B(src + (size_t)t * 8,         dst + (size_t)t * 8);
      stage16B(src + (size_t)(t + 256) * 8, dst + (size_t)(t + 256) * 8);
    }
    if (cb + 1 < N_TOK / 32)
      __builtin_prefetch(Kb + (size_t)(cb0 + 32) * DKV + t * 16, 0, 0);
    async_stage_fence();
    __syncthreads();

    v8f c0 = v8f_zero(), c1 = v8f_zero();
#pragma unroll
    for (int kk = 0; kk < 4; ++kk) {
      const unsigned short* b0 = &Kc[lc][kk * 32 + half * 8];
      const unsigned short* b1 = &Kc[16 + lc][kk * 32 + half * 8];
      c0 = WMMA_BF16(aq[kk], ld16(b0, b0 + 16), c0);
      c1 = WMMA_BF16(aq[kk], ld16(b1, b1 + 16), c1);
    }
    // online (max, sumexp) update — each lane owns 2 columns of 8 rows
#pragma unroll
    for (int r = 0; r < 8; ++r) {
      float s0 = c0[r] * scale, s1 = c1[r] * scale;
      float nm = fmaxf(m[r], fmaxf(s0, s1));
      s[r] = s[r] * __expf(m[r] - nm) + __expf(s0 - nm) + __expf(s1 - nm);
      m[r] = nm;
    }
  }

  // combine across the 16 lanes that share each row (xor stays in half-wave)
#pragma unroll
  for (int r = 0; r < 8; ++r) {
#pragma unroll
    for (int mask = 1; mask < 16; mask <<= 1) {
      float om = __shfl_xor(m[r], mask, 32);
      float os = __shfl_xor(s[r], mask, 32);
      float nm = fmaxf(m[r], om);
      s[r] = s[r] * __expf(m[r] - nm) + os * __expf(om - nm);
      m[r] = nm;
    }
  }
  if (lc == 0) {
#pragma unroll
    for (int r = 0; r < 8; ++r) {
      int row = q0 + r + 8 * half;
      mrow[row] = m[r];
      drow[row] = s[r];
    }
  }
}

// ---------------------------------------------------------------------------
// Kernel 3: pass 2 — recompute score tiles, write normalized probabilities
// once (non-temporal, 256 MB), and accumulate O = P @ V with WMMA.
// ---------------------------------------------------------------------------
__global__ __launch_bounds__(256) void attn_kernel(
    const unsigned short* __restrict__ Qb, const unsigned short* __restrict__ Kb,
    const unsigned short* __restrict__ Vb,
    const float* __restrict__ mrow, const float* __restrict__ drow,
    float* __restrict__ O, float* __restrict__ Att)
{
  __shared__ __align__(16) unsigned short Kc[32][DKV];   // K chunk, row-major
  __shared__ __align__(16) unsigned short Vt[DKV][32];   // V chunk, transposed
  __shared__ __align__(16) unsigned short Pw[8][16][32]; // per-wave P patch

  const int t    = threadIdx.x;
  const int wave = t >> 5, lane = t & 31;
  const int half = lane >> 4, lc = lane & 15;
  const int q0   = blockIdx.x * 128 + wave * 16;
  const float scale = 0.08838834764831845f;

  v16bf aq[4];
#pragma unroll
  for (int kk = 0; kk < 4; ++kk) {
    const unsigned short* p = Qb + (size_t)(q0 + lc) * DKV + kk * 32 + half * 8;
    aq[kk] = ld16(p, p + 16);
  }

  float mr[8], di[8];
#pragma unroll
  for (int r = 0; r < 8; ++r) {
    int row = q0 + r + 8 * half;
    mr[r] = mrow[row];
    di[r] = 1.0f / drow[row];
  }

  v8f o[8];
#pragma unroll
  for (int i = 0; i < 8; ++i) o[i] = v8f_zero();

  for (int cb = 0; cb < N_TOK / 32; ++cb) {
    const int cb0 = cb * 32;
    __syncthreads();
    {
      const unsigned short* src = Kb + (size_t)cb0 * DKV;
      unsigned short* dst = &Kc[0][0];
      stage16B(src + (size_t)t * 8,         dst + (size_t)t * 8);
      stage16B(src + (size_t)(t + 256) * 8, dst + (size_t)(t + 256) * 8);
#pragma unroll
      for (int i = 0; i < 16; ++i) {        // V staged transposed (scatter)
        int idx = t + 256 * i;
        int r = idx >> 7, c = idx & 127;
        Vt[c][r] = Vb[(size_t)(cb0 + r) * DKV + c];
      }
    }
    if (cb + 1 < N_TOK / 32) {
      __builtin_prefetch(Kb + (size_t)(cb0 + 32) * DKV + t * 16, 0, 0);
      __builtin_prefetch(Vb + (size_t)(cb0 + 32) * DKV + t * 16, 0, 0);
    }
    async_stage_fence();
    __syncthreads();

    // --- scores ---
    v8f c0 = v8f_zero(), c1 = v8f_zero();
#pragma unroll
    for (int kk = 0; kk < 4; ++kk) {
      const unsigned short* b0 = &Kc[lc][kk * 32 + half * 8];
      const unsigned short* b1 = &Kc[16 + lc][kk * 32 + half * 8];
      c0 = WMMA_BF16(aq[kk], ld16(b0, b0 + 16), c0);
      c1 = WMMA_BF16(aq[kk], ld16(b1, b1 + 16), c1);
    }

    // --- probabilities: NT store to d_out + stage bf16 into LDS for AV ---
#pragma unroll
    for (int r = 0; r < 8; ++r) {
      int row = q0 + r + 8 * half;
      float p0 = __expf(c0[r] * scale - mr[r]) * di[r];
      float p1 = __expf(c1[r] * scale - mr[r]) * di[r];
      __builtin_nontemporal_store(p0, &Att[(size_t)row * N_TOK + cb0 + lc]);
      __builtin_nontemporal_store(p1, &Att[(size_t)row * N_TOK + cb0 + 16 + lc]);
      Pw[wave][r + 8 * half][lc]      = f2bf_u(p0);
      Pw[wave][r + 8 * half][16 + lc] = f2bf_u(p1);
    }
    __syncthreads();   // LDS transpose barrier (also protects Kc/Vt reuse)

    // --- O += P(16x32) @ V(32x128) ---
    const unsigned short* pa0 = &Pw[wave][lc][half * 8];
    const unsigned short* pa1 = &Pw[wave][lc][16 + half * 8];
    v16bf pa = ld16(pa0, pa1);
#pragma unroll
    for (int vt = 0; vt < 8; ++vt) {
      const unsigned short* bp = &Vt[vt * 16 + lc][half * 8];
      o[vt] = WMMA_BF16(pa, ld16(bp, bp + 16), o[vt]);
    }
  }

#pragma unroll
  for (int vt = 0; vt < 8; ++vt)
#pragma unroll
    for (int r = 0; r < 8; ++r) {
      int row = q0 + r + 8 * half;
      O[(size_t)row * DKV + vt * 16 + lc] = o[vt][r];
    }
}

// ---------------------------------------------------------------------------
extern "C" void kernel_launch(void* const* d_in, const int* in_sizes, int n_in,
                              void* d_out, int out_size, void* d_ws, size_t ws_size,
                              hipStream_t stream) {
  const float* X  = (const float*)d_in[0];
  const float* Wq = (const float*)d_in[1];
  const float* Wk = (const float*)d_in[2];
  const float* Wv = (const float*)d_in[3];

  float* O   = (float*)d_out;                       // [8192,128]
  float* Att = (float*)d_out + (size_t)N_TOK * DKV; // [8192,8192]

  unsigned short* Qb = (unsigned short*)d_ws;                 // bf16 Q
  unsigned short* Kb = Qb + (size_t)N_TOK * DKV;              // bf16 K
  unsigned short* Vb = Kb + (size_t)N_TOK * DKV;              // bf16 V
  float* mrow = (float*)(Vb + (size_t)N_TOK * DKV);           // row max
  float* drow = mrow + N_TOK;                                 // row sumexp

  dim3 gproj(N_TOK / 128, 3);
  proj_kernel<<<gproj, 256, 0, stream>>>(X, Wq, Wk, Wv, Qb, Kb, Vb);
  stats_kernel<<<N_TOK / 128, 256, 0, stream>>>(Qb, Kb, mrow, drow);
  attn_kernel<<<N_TOK / 128, 256, 0, stream>>>(Qb, Kb, Vb, mrow, drow, O, Att);
}
